// JointMMwithRel_7043746365896
// MI455X (gfx1250) — matrix operand order repaired and checked by reading, not compiled, not run on hardware
//
#include <hip/hip_runtime.h>
#include <hip/hip_bf16.h>
#include <math.h>

// ---------------------------------------------------------------------------
// Types for CDNA5 WMMA (wave32, 16x16x32 bf16 -> f32)
// ---------------------------------------------------------------------------
typedef __attribute__((ext_vector_type(16))) __bf16 v16bf;
typedef __attribute__((ext_vector_type(8)))  __bf16 v8bf;
typedef __attribute__((ext_vector_type(8)))  float  v8f;

#define OUT_F  1
#define OUT_B  2
#define OUT_VT 4

__device__ __forceinline__ v8f wmma_bf16(v16bf a, v16bf b, v8f c) {
  return __builtin_amdgcn_wmma_f32_16x16x32_bf16(false, a, false, b, (short)0, c,
                                                 false, false);
}

// A-operand fragment (16x32 bf16, MxK). Lane l: row = l&15, K elems
// {koff..koff+15}, koff = (l>>4)*16 already folded into p.
__device__ __forceinline__ v16bf mk_a(const __bf16* p) {
  return *(const v16bf*)p;
}

// B-operand fragment where the logical B column is a K-contiguous row in
// memory (we compute X*W^T): lane holds 16 contiguous bf16.
__device__ __forceinline__ v16bf mk_b(const __bf16* p) {
  return *(const v16bf*)p;
}

// ---------------------------------------------------------------------------
// Generic GEMM: C[M,N] = A_bf16[M,K] * W_f32[N,K(ldw)]^T  (+bias)
// Each wave computes a 32x64 C tile (2x4 fragments, 8 WMMAs / K=32 step).
// All four B fragments are loaded+converted before the WMMA burst so their
// live ranges stay distinct (avoids WMMA->VALU WAR hazard nops).
// Assumptions: N%64==0, K%32==0.
// ---------------------------------------------------------------------------
typedef __attribute__((ext_vector_type(4))) float v4f;

__global__ void __launch_bounds__(256) gemm_wmma(
    const __bf16* __restrict__ A, const float* __restrict__ W,
    const float* __restrict__ bias, float* __restrict__ Cf,
    __bf16* __restrict__ Cb, int M, int N, int K, int ldw, int mode, int Lk,
    int accum) {
  int lane = threadIdx.x & 31;
  int wave = (int)((blockIdx.x * blockDim.x + threadIdx.x) >> 5);
  int mt = (M + 31) >> 5, nt = N >> 6;
  if (wave >= mt * nt) return;
  int m0 = (wave % mt) << 5;
  int n0 = (wave / mt) << 6;
  int l15 = lane & 15, half = lane >> 4;

  int rowA0 = m0 + l15;       if (rowA0 > M - 1) rowA0 = M - 1;
  int rowA1 = m0 + 16 + l15;  if (rowA1 > M - 1) rowA1 = M - 1;
  const __bf16* pa0 = A + (size_t)rowA0 * K + half * 16;
  const __bf16* pa1 = A + (size_t)rowA1 * K + half * 16;
  const float* pw[4];
#pragma unroll
  for (int ni = 0; ni < 4; ni++)
    pw[ni] = W + (size_t)(n0 + ni * 16 + l15) * ldw + half * 16;

  v8f acc[2][4];
#pragma unroll
  for (int mi = 0; mi < 2; mi++)
#pragma unroll
    for (int ni = 0; ni < 4; ni++)
#pragma unroll
      for (int i = 0; i < 8; i++) acc[mi][ni][i] = 0.f;

  for (int k0 = 0; k0 < K; k0 += 32) {
    __builtin_prefetch((const void*)(pw[0] + k0 + 512), 0, 1);
    // Stage 1: issue every load for this K step (A + 4x16 f32 of W).
    v16bf a0 = mk_a(pa0 + k0);
    v16bf a1 = mk_a(pa1 + k0);
    v4f wf[4][4];
#pragma unroll
    for (int ni = 0; ni < 4; ni++)
#pragma unroll
      for (int j = 0; j < 4; j++)
        wf[ni][j] = *(const v4f*)(pw[ni] + k0 + j * 4);
    // Stage 2: convert to bf16 fragments (distinct live ranges).
    v16bf b[4];
#pragma unroll
    for (int ni = 0; ni < 4; ni++)
#pragma unroll
      for (int j = 0; j < 4; j++)
#pragma unroll
        for (int e = 0; e < 4; e++)
          b[ni][j * 4 + e] = (__bf16)wf[ni][j][e];
    // Stage 3: WMMA burst.
#pragma unroll
    for (int ni = 0; ni < 4; ni++) {
      acc[0][ni] = wmma_bf16(a0, b[ni], acc[0][ni]);
      acc[1][ni] = wmma_bf16(a1, b[ni], acc[1][ni]);
    }
  }

#pragma unroll
  for (int mi = 0; mi < 2; mi++) {
#pragma unroll
    for (int ni = 0; ni < 4; ni++) {
      int nbase = n0 + ni * 16;
      int mbase = m0 + mi * 16;
      int n = nbase + l15;
      float bv = bias ? bias[n] : 0.0f;
#pragma unroll
      for (int r = 0; r < 8; r++) {
        int m = mbase + r + half * 8;
        if (m < M) {
          float v = acc[mi][ni][r] + bv;
          size_t idx = (size_t)m * N + n;
          if (accum) v += Cf[idx];
          if (mode & OUT_F) Cf[idx] = v;
          if (mode & OUT_B) Cb[idx] = (__bf16)v;
          if (mode & OUT_VT) {
            int bb = m / Lk, lk = m - bb * Lk;
            int h = n / 768, d = n - h * 768;
            // vT[b][h][d][k], k-stride padded to 128
            Cb[(((size_t)(bb * 8 + h) * 768) + d) * 128 + lk] = (__bf16)v;
          }
        }
      }
    }
  }
}

// ---------------------------------------------------------------------------
// Fused attention: per (b, h, 16-row q tile), one wave.
// scores = Q*K^T/sqrt(768) -> softmax (LDS) -> ctx = P*V (Vt is [b,h,d,k]).
// Templated on NKT (= ceil(Lk/16)) and KCH (= ceil(Lk/32)) so the WMMA loops
// fully unroll and loads batch ahead of the waits.
// ---------------------------------------------------------------------------
template <int NKT, int KCH>
__global__ void __launch_bounds__(32) attn_kernel(
    const __bf16* __restrict__ Q, const __bf16* __restrict__ Kb,
    const __bf16* __restrict__ Vt, __bf16* __restrict__ ctx, int Lq, int Lk,
    float scale) {
  __shared__ __align__(16) float  s_sc[16][128];
  __shared__ __align__(16) __bf16 s_pb[16][128];

  int lane = threadIdx.x;
  int nqt = (Lq + 15) >> 4;
  int bid = blockIdx.x;
  int qt = bid % nqt;
  int h = (bid / nqt) & 7;
  int b = bid / (nqt * 8);
  int l15 = lane & 15, half = lane >> 4;

  int qrow = qt * 16 + l15;
  if (qrow > Lq - 1) qrow = Lq - 1;
  const __bf16* qp = Q + ((size_t)(b * Lq + qrow)) * 6144 + h * 768 + half * 16;

  const __bf16* kbase[NKT];
#pragma unroll
  for (int kt = 0; kt < NKT; ++kt) {
    int kr = kt * 16 + l15;
    if (kr > Lk - 1) kr = Lk - 1;
    kbase[kt] = Kb + ((size_t)(b * Lk + kr)) * 6144 + h * 768 + half * 16;
  }

  v8f sacc[NKT];
#pragma unroll
  for (int t = 0; t < NKT; t++)
#pragma unroll
    for (int i = 0; i < 8; i++) sacc[t][i] = 0.f;

  for (int k0 = 0; k0 < 768; k0 += 32) {
    v16bf a = mk_a(qp + k0);
    v16bf kb[NKT];
#pragma unroll
    for (int kt = 0; kt < NKT; ++kt) kb[kt] = mk_b(kbase[kt] + k0);
#pragma unroll
    for (int kt = 0; kt < NKT; ++kt)
      sacc[kt] = wmma_bf16(a, kb[kt], sacc[kt]);
  }

  // scores -> LDS (C layout: row = r + 8*half, col = kt*16 + l15)
#pragma unroll
  for (int kt = 0; kt < NKT; ++kt)
#pragma unroll
    for (int r = 0; r < 8; r++)
      s_sc[r + half * 8][kt * 16 + l15] = sacc[kt][r] * scale;
  __syncthreads();

  if (lane < 16) {
    const int kpad = KCH * 32;
    float mx = -3.0e38f;
    for (int c = 0; c < Lk; c++) mx = fmaxf(mx, s_sc[lane][c]);
    float sum = 0.f;
    for (int c = 0; c < Lk; c++) {
      float e = __expf(s_sc[lane][c] - mx);
      s_sc[lane][c] = e;
      sum += e;
    }
    float inv = 1.0f / sum;
    for (int c = 0; c < kpad; c++)
      s_pb[lane][c] = (c < Lk) ? (__bf16)(s_sc[lane][c] * inv) : (__bf16)0.0f;
  }
  __syncthreads();

  // ctx = P * V ; Vt[b][h][d][k] gives K-contiguous B-fragment rows.
  v16bf pa[KCH];
#pragma unroll
  for (int kc = 0; kc < KCH; ++kc)
    pa[kc] = mk_a(&s_pb[l15][kc * 32 + half * 16]);  // ds loads

  for (int dt = 0; dt < 48; ++dt) {
    v8f acc;
#pragma unroll
    for (int i = 0; i < 8; i++) acc[i] = 0.f;
    int d = dt * 16 + l15;
    const __bf16* vp = Vt + (((size_t)(b * 8 + h) * 768) + d) * 128 + half * 16;
    v16bf vb[KCH];
#pragma unroll
    for (int kc = 0; kc < KCH; ++kc) vb[kc] = mk_b(vp + kc * 32);
#pragma unroll
    for (int kc = 0; kc < KCH; ++kc) acc = wmma_bf16(pa[kc], vb[kc], acc);
#pragma unroll
    for (int r = 0; r < 8; r++) {
      int q = qt * 16 + r + half * 8;
      if (q < Lq)
        ctx[((size_t)(b * Lq + q)) * 6144 + h * 768 + dt * 16 + l15] =
            (__bf16)acc[r];
    }
  }
}

// ---------------------------------------------------------------------------
// Residual + LayerNorm over 768, one block per row, emits f32 + bf16.
// ---------------------------------------------------------------------------
__global__ void __launch_bounds__(256) ln_kernel(
    const float* __restrict__ x, const float* __restrict__ res,
    const float* __restrict__ g, const float* __restrict__ bta,
    float* __restrict__ outf, __bf16* __restrict__ outb, int M) {
  __shared__ float red[256];
  int row = blockIdx.x;
  if (row >= M) return;
  int t = threadIdx.x;
  const float* xr = x + (size_t)row * 768;
  const float* rr = res + (size_t)row * 768;
  float v0 = xr[t] + rr[t];
  float v1 = xr[t + 256] + rr[t + 256];
  float v2 = xr[t + 512] + rr[t + 512];
  red[t] = v0 + v1 + v2;
  __syncthreads();
  for (int st = 128; st > 0; st >>= 1) {
    if (t < st) red[t] += red[t + st];
    __syncthreads();
  }
  float mu = red[0] * (1.0f / 768.0f);
  __syncthreads();
  float d0 = v0 - mu, d1 = v1 - mu, d2 = v2 - mu;
  red[t] = d0 * d0 + d1 * d1 + d2 * d2;
  __syncthreads();
  for (int st = 128; st > 0; st >>= 1) {
    if (t < st) red[t] += red[t + st];
    __syncthreads();
  }
  float inv = rsqrtf(red[0] * (1.0f / 768.0f) + 1e-5f);
  float o0 = d0 * inv * g[t] + bta[t];
  float o1 = d1 * inv * g[t + 256] + bta[t + 256];
  float o2 = d2 * inv * g[t + 512] + bta[t + 512];
  size_t base = (size_t)row * 768;
  outf[base + t] = o0;
  outf[base + t + 256] = o1;
  outf[base + t + 512] = o2;
  outb[base + t] = (__bf16)o0;
  outb[base + t + 256] = (__bf16)o1;
  outb[base + t + 512] = (__bf16)o2;
}

// ---------------------------------------------------------------------------
// f32 -> bf16 elementwise convert
// ---------------------------------------------------------------------------
__global__ void __launch_bounds__(256) cvt_bf16(const float* __restrict__ in,
                                                __bf16* __restrict__ out,
                                                int n) {
  int i = blockIdx.x * 256 + threadIdx.x;
  if (i < n) out[i] = (__bf16)in[i];
}

// ---------------------------------------------------------------------------
// Host orchestration
// ---------------------------------------------------------------------------
extern "C" void kernel_launch(void* const* d_in, const int* in_sizes, int n_in,
                              void* d_out, int out_size, void* d_ws,
                              size_t ws_size, hipStream_t stream) {
  (void)in_sizes; (void)n_in; (void)out_size; (void)ws_size;

  // --- workspace carve-up (256B aligned) ---
  size_t off = 0;
  auto alloc = [&](size_t bytes) -> void* {
    void* p = (void*)((char*)d_ws + off);
    off += (bytes + 255) & ~(size_t)255;
    return p;
  };
  __bf16* textb = (__bf16*)alloc((size_t)2048 * 768 * 2);
  __bf16* imgfb = (__bf16*)alloc((size_t)784 * 2048 * 2);
  float*  img_f = (float*) alloc((size_t)784 * 768 * 4);
  __bf16* img_b = (__bf16*)alloc((size_t)784 * 768 * 2);
  float*  hel_f = (float*) alloc((size_t)784 * 768 * 4);
  __bf16* hel_b = (__bf16*)alloc((size_t)784 * 768 * 2);
  float*  tA_f  = (float*) alloc((size_t)2048 * 768 * 4);
  __bf16* tA_b  = (__bf16*)alloc((size_t)2048 * 768 * 2);
  float*  tB_f  = (float*) alloc((size_t)2048 * 768 * 4);
  __bf16* tB_b  = (__bf16*)alloc((size_t)2048 * 768 * 2);
  __bf16* qp    = (__bf16*)alloc((size_t)2048 * 6144 * 2);
  __bf16* kp    = (__bf16*)alloc((size_t)2048 * 6144 * 2);
  __bf16* vT    = (__bf16*)alloc((size_t)16 * 8 * 768 * 128 * 2);
  __bf16* ctx   = (__bf16*)alloc((size_t)2048 * 6144 * 2);
  float*  fco   = (float*) alloc((size_t)2048 * 768 * 4);

  const float scale = 1.0f / sqrtf(768.0f);

  auto launch_gemm = [&](const __bf16* A, const float* W, const float* bias,
                         float* Cf, __bf16* Cb, int M, int N, int K, int ldw,
                         int mode, int Lk, int accum) {
    int mt = (M + 31) / 32, nt = N / 64;
    long long waves = (long long)mt * nt;
    int blocks = (int)((waves * 32 + 255) / 256);
    gemm_wmma<<<dim3(blocks), dim3(256), 0, stream>>>(A, W, bias, Cf, Cb, M, N,
                                                      K, ldw, mode, Lk, accum);
  };

  auto run_mha = [&](const __bf16* qb, const float* qf, int Lq,
                     const __bf16* kb, int Lk, int pb, float* of, __bf16* ob) {
    const float* wq = (const float*)d_in[pb + 0];
    const float* wk = (const float*)d_in[pb + 1];
    const float* wv = (const float*)d_in[pb + 2];
    const float* fc = (const float*)d_in[pb + 3];
    const float* g  = (const float*)d_in[pb + 4];
    const float* bt = (const float*)d_in[pb + 5];
    int Mq = 16 * Lq, Mk = 16 * Lk;
    launch_gemm(qb, wq, nullptr, nullptr, qp, Mq, 6144, 768, 768, OUT_B, 0, 0);
    launch_gemm(kb, wk, nullptr, nullptr, kp, Mk, 6144, 768, 768, OUT_B, 0, 0);
    launch_gemm(kb, wv, nullptr, nullptr, vT, Mk, 6144, 768, 768, OUT_VT, Lk, 0);
    int nqt = (Lq + 15) / 16;
    dim3 agrid(16 * 8 * nqt), ablk(32);
    if (Lk == 128)
      attn_kernel<8, 4><<<agrid, ablk, 0, stream>>>(qp, kp, vT, ctx, Lq, Lk,
                                                    scale);
    else
      attn_kernel<4, 2><<<agrid, ablk, 0, stream>>>(qp, kp, vT, ctx, Lq, Lk,
                                                    scale);
    launch_gemm(ctx, fc, nullptr, fco, nullptr, Mq, 768, 6144, 6144, OUT_F, 0, 0);
    ln_kernel<<<dim3(Mq), dim3(256), 0, stream>>>(fco, qf, g, bt, of, ob, Mq);
  };

  const float* text_f = (const float*)d_in[0];
  const float* img_feat = (const float*)d_in[1];

  // zero vT once per call so padded k-lanes multiply probs==0 cleanly
  hipMemsetAsync(vT, 0, (size_t)16 * 8 * 768 * 128 * 2, stream);

  // bf16 conversions of the raw inputs
  {
    int n = 16 * 128 * 768;
    cvt_bf16<<<dim3((n + 255) / 256), dim3(256), 0, stream>>>(text_f, textb, n);
    n = 16 * 49 * 2048;
    cvt_bf16<<<dim3((n + 255) / 256), dim3(256), 0, stream>>>(img_feat, imgfb, n);
  }

  // img_rep = img_feat @ align_w^T + align_b
  launch_gemm(imgfb, (const float*)d_in[32], (const float*)d_in[33], img_f,
              img_b, 784, 768, 2048, 2048, OUT_F | OUT_B, 0, 0);

  // helighted = MHA(img_rep, text, text; helight)          params @2
  run_mha(img_b, img_f, 49, textb, 128, 2, hel_f, hel_b);
  // t1 = MHA(helighted, text; attn1)                       params @8
  run_mha(hel_b, hel_f, 49, textb, 128, 8, tA_f, tA_b);
  // t2 = MHA(img_rep, t1; attn2)                           params @14
  run_mha(img_b, img_f, 49, tA_b, 49, 14, tB_f, tB_b);
  // t3 = MHA(text, t2; attn3)                              params @20
  run_mha(textb, text_f, 128, tB_b, 49, 20, tA_f, tA_b);
  // t4 = MHA(img_rep, t3; attn1)
  run_mha(img_b, img_f, 49, tA_b, 128, 8, tB_f, tB_b);
  // t5 = MHA(helighted, t4; attn2)
  run_mha(hel_b, hel_f, 49, tB_b, 49, 14, tA_f, tA_b);
  // t6 = MHA(text, t5; attn3)
  run_mha(textb, text_f, 128, tA_b, 49, 20, tB_f, tB_b);
  // t7 = MHA(t6, t6; attn7)                                params @26
  run_mha(tB_b, tB_f, 128, tB_b, 128, 26, tA_f, tA_b);

  // fused = concat(t7, text) @ mlp1_w^T + mlp1_b  (two accumulating GEMMs;
  // W is [768,1536] row-major: first half cols for t7, second half for text)
  const float* mlp_w = (const float*)d_in[34];
  const float* mlp_b = (const float*)d_in[35];
  launch_gemm(tA_b, mlp_w, mlp_b, (float*)d_out, nullptr, 2048, 768, 768, 1536,
              OUT_F, 0, 0);
  launch_gemm(textb, mlp_w + 768, nullptr, (float*)d_out, nullptr, 2048, 768,
              768, 1536, OUT_F, 0, 1);
}
